// generator_37649683317346
// MI455X (gfx1250) — matrix operand order repaired and checked by reading, not compiled, not run on hardware
//
#include <hip/hip_runtime.h>

typedef __bf16 bf16;
typedef __attribute__((ext_vector_type(8)))  bf16  v8bf;
typedef __attribute__((ext_vector_type(16))) bf16  v16bf;
typedef __attribute__((ext_vector_type(8)))  float v8f;

// ---------- helpers ----------
__device__ __forceinline__ bf16 f2bf(float f) {
  union { float f; unsigned u; } v; v.f = f;
  unsigned r = v.u + 0x7FFFu + ((v.u >> 16) & 1u);   // round-to-nearest-even
  unsigned short h = (unsigned short)(r >> 16);
  return __builtin_bit_cast(bf16, h);
}
__device__ __forceinline__ float bf2f(bf16 b) {
  unsigned short h = __builtin_bit_cast(unsigned short, b);
  union { unsigned u; float f; } v; v.u = ((unsigned)h) << 16;
  return v.f;
}
__device__ __forceinline__ float sigm(float x) { return 1.0f / (1.0f + __expf(-x)); }

__device__ __forceinline__ v16bf load_a_frag(const bf16* ap) {
  v8bf alo = *(const v8bf*)(ap);
  v8bf ahi = *(const v8bf*)(ap + 16);
  return __builtin_shufflevector(alo, ahi, 0,1,2,3,4,5,6,7,8,9,10,11,12,13,14,15);
}

// ---------- small WMMA GEMM (one 16x16 tile per wave; used for M=32 recurrent GEMMs) ----------
// A: bf16 row-major [M][K]; Bpack: [(K/32)][N/16][32 lanes][16 halves]
__global__ void gemm_bf16_wmma(const bf16* __restrict__ A, const bf16* __restrict__ Bp,
                               const float* __restrict__ bias,
                               float* __restrict__ C, bf16* __restrict__ Cbf,
                               int M, int N, int K, int act, int permM)
{
  const int lane = threadIdx.x & 31;
  const int ntN  = N >> 4;
  const int ntM  = M >> 4;
  int tile = blockIdx.x * (blockDim.x >> 5) + (threadIdx.x >> 5);
  if (tile >= ntM * ntN) return;                 // wave-uniform exit
  const int mt = tile / ntN;
  const int nt = tile - mt * ntN;

  const int row   = (mt << 4) + (lane & 15);
  const int khalf = (lane >> 4) << 3;            // 0 or 8
  const int nk    = K >> 5;

  const bf16* ap = A + (size_t)row * K + khalf;
  const bf16* bp = Bp + ((size_t)nt * 32 + lane) * 16;
  const size_t bstride = (size_t)ntN * 32 * 16;

  v8f acc = {};
  for (int kt = 0; kt < nk; ++kt) {
    v16bf a = load_a_frag(ap);
    v16bf b = *(const v16bf*)(bp);
    if (kt + 1 < nk) __builtin_prefetch((const void*)(bp + bstride), 0, 1);
    acc = __builtin_amdgcn_wmma_f32_16x16x32_bf16(false, a, false, b, (short)0, acc, false, false);
    ap += 32;
    bp += bstride;
  }

  const int col   = (nt << 4) + (lane & 15);
  const float bv  = bias ? bias[col] : 0.0f;
  const int mbase = (mt << 4) + khalf;
#pragma unroll
  for (int i = 0; i < 8; ++i) {
    int m = mbase + i;
    float v = acc[i] + bv;
    if (act) v = tanhf(v);
    int orow = permM ? ((m & 31) * permM + (m >> 5)) : m;
    C[(size_t)orow * N + col] = v;
    if (Cbf) Cbf[(size_t)orow * N + col] = f2bf(v);
  }
}

// ---------- big WMMA GEMM: 128x64 macro-tile per block, B staged via async-to-LDS ----------
// grid = (N/64, Mpad/128), 256 threads (8 waves). Wave w owns M-tile (blockIdx.y*8 + w)
// and computes 4 N-tiles; the 4 KB of B fragments per K-step are fetched once per block
// with GLOBAL_LOAD_ASYNC_TO_LDS_B128 (ASYNCcnt) and consumed with ds_load_b128.
__global__ void gemm_bf16_wmma_big(const bf16* __restrict__ A, const bf16* __restrict__ Bp,
                                   const float* __restrict__ bias, float* __restrict__ C,
                                   int Mreal, int N, int K, int permM)
{
  __shared__ __align__(16) bf16 sB[4 * 32 * 16];   // 4 KB: 4 N-tile fragments
  const int tid  = threadIdx.x;
  const int lane = tid & 31;
  const int wave = tid >> 5;                       // 0..7
  const int ntN  = N >> 4;
  const int ntBase = blockIdx.x << 2;              // 4 N-tiles per block
  const int mt   = (blockIdx.y << 3) + wave;
  const int nk   = K >> 5;

  const int row   = (mt << 4) + (lane & 15);
  const int khalf = (lane >> 4) << 3;
  const bf16* ap  = A + (size_t)row * K + khalf;

  // async staging: thread tid moves one 16B chunk of fragment q per K-step
  const int q  = tid >> 6;                          // N-tile 0..3
  const int ch = tid & 63;                          // 16B chunk within 1 KB fragment
  unsigned long long gsrc =
      (unsigned long long)(size_t)(Bp + ((size_t)(ntBase + q) * 512) + (size_t)ch * 8);
  const unsigned long long gstride = (unsigned long long)ntN * 512 * sizeof(bf16);
  const unsigned ldsoff = (unsigned)(size_t)(&sB[q * 512 + ch * 8]);

  v8f acc0 = {}, acc1 = {}, acc2 = {}, acc3 = {};
  for (int kt = 0; kt < nk; ++kt) {
    asm volatile("global_load_async_to_lds_b128 %0, %1, off"
                 :: "v"(ldsoff), "v"(gsrc) : "memory");
    asm volatile("s_wait_asynccnt 0x0" ::: "memory");
    __syncthreads();
    v16bf a  = load_a_frag(ap);
    v16bf b0 = *(const v16bf*)(&sB[0 * 512 + lane * 16]);
    v16bf b1 = *(const v16bf*)(&sB[1 * 512 + lane * 16]);
    v16bf b2 = *(const v16bf*)(&sB[2 * 512 + lane * 16]);
    v16bf b3 = *(const v16bf*)(&sB[3 * 512 + lane * 16]);
    acc0 = __builtin_amdgcn_wmma_f32_16x16x32_bf16(false, a, false, b0, (short)0, acc0, false, false);
    acc1 = __builtin_amdgcn_wmma_f32_16x16x32_bf16(false, a, false, b1, (short)0, acc1, false, false);
    acc2 = __builtin_amdgcn_wmma_f32_16x16x32_bf16(false, a, false, b2, (short)0, acc2, false, false);
    acc3 = __builtin_amdgcn_wmma_f32_16x16x32_bf16(false, a, false, b3, (short)0, acc3, false, false);
    __syncthreads();                                // protect sB before next overwrite
    ap += 32;
    gsrc += gstride;
  }

  const int mbase = (mt << 4) + khalf;
#pragma unroll
  for (int t = 0; t < 4; ++t) {
    v8f acc = (t == 0) ? acc0 : (t == 1) ? acc1 : (t == 2) ? acc2 : acc3;
    const int col  = ((ntBase + t) << 4) + (lane & 15);
    const float bv = bias ? bias[col] : 0.0f;
#pragma unroll
    for (int i = 0; i < 8; ++i) {
      int m = mbase + i;
      if (m < Mreal) {
        int orow = permM ? ((m & 31) * permM + (m >> 5)) : m;
        C[(size_t)orow * N + col] = acc[i] + bv;
      }
    }
  }
}

// ---------- weight pack: W[N][K] fp32 -> bf16 WMMA B-fragments ----------
__global__ void pack_w(const float* __restrict__ W, bf16* __restrict__ P, int N, int K) {
  int idx = blockIdx.x * blockDim.x + threadIdx.x;
  if (idx >= N * K) return;
  int i16  = idx & 15;
  int lane = (idx >> 4) & 31;
  int ntN  = N >> 4;
  int nt   = (idx >> 9) % ntN;
  int kt   = (idx >> 9) / ntN;
  int k = (kt << 5) + ((lane >> 4) << 3) + (i16 < 8 ? i16 : 8 + i16 - 8 + 8);
  int n = (nt << 4) + (lane & 15);
  P[idx] = f2bf(W[(size_t)n * K + k]);
}

// ---------- small utility kernels ----------
__global__ void vec_add2(const float* a, const float* b, float* o, int n) {
  int i = blockIdx.x * blockDim.x + threadIdx.x;
  if (i < n) o[i] = a[i] + b[i];
}
__global__ void f32_to_bf16(const float* a, bf16* o, int n) {
  int i = blockIdx.x * blockDim.x + threadIdx.x;
  if (i < n) o[i] = f2bf(a[i]);
}
__global__ void zero_f32(float* p, int n) {
  int i = blockIdx.x * blockDim.x + threadIdx.x;
  if (i < n) p[i] = 0.0f;
}
__global__ void zero_bf16(bf16* p, int n) {
  int i = blockIdx.x * blockDim.x + threadIdx.x;
  if (i < n) p[i] = f2bf(0.0f);
}
__global__ void i2f(const int* a, float* o, int n) {
  int i = blockIdx.x * blockDim.x + threadIdx.x;
  if (i < n) o[i] = (float)a[i];
}
__global__ void embed_gather(const float* __restrict__ emb, const int* __restrict__ sent,
                             bf16* __restrict__ o, int n) {
  int idx = blockIdx.x * blockDim.x + threadIdx.x;
  if (idx >= n) return;
  int e = idx & 511;        // E = 512
  int r = idx >> 9;
  int b = r & 31;
  int t = r >> 5;
  int tok = sent[b * 64 + t];
  o[idx] = f2bf(emb[(size_t)tok * 512 + e]);
}

// ---------- fused LSTM elementwise (gate order i,f,g,o) ----------
__global__ void lstm_elem(const float* __restrict__ xg, const float* __restrict__ hg,
                          float* __restrict__ c, float* __restrict__ hf,
                          bf16* __restrict__ hbf, bf16* __restrict__ hall) {
  int idx = blockIdx.x * blockDim.x + threadIdx.x;
  if (idx >= 32 * 1024) return;
  int b = idx >> 10, j = idx & 1023;
  int base = b << 12;
  float gi = xg[base + j]        + hg[base + j];
  float gf = xg[base + 1024 + j] + hg[base + 1024 + j];
  float gg = xg[base + 2048 + j] + hg[base + 2048 + j];
  float go = xg[base + 3072 + j] + hg[base + 3072 + j];
  float cn = sigm(gf) * c[idx] + sigm(gi) * tanhf(gg);
  float hn = sigm(go) * tanhf(cn);
  c[idx]  = cn;
  hf[idx] = hn;
  bf16 hb = f2bf(hn);
  hbf[idx] = hb;
  if (hall) hall[idx] = hb;
}

// ---------- per-step attention + cell gating (one block per batch row) ----------
__global__ void attn_step(const bf16* __restrict__ hall1, const float* __restrict__ pp,
                          const float* __restrict__ cur,
                          const float* __restrict__ attW, const float* __restrict__ attB,
                          const float* __restrict__ gateW, const float* __restrict__ gateB,
                          const int* __restrict__ len1,
                          const float* __restrict__ h2, float* __restrict__ c2) {
  const int L = 63, Hd = 1024;
  int b = blockIdx.x, tid = threadIdx.x;
  __shared__ float red[256];
  __shared__ float w[64];
  __shared__ float gsh;
  int lim = len1[b] - 1;
  for (int t = 0; t < L; ++t) {
    float p = 0.0f;
    const float* ppr = pp + ((size_t)(t * 32 + b)) * Hd;
    const float* crr = cur + (size_t)b * Hd;
    for (int j = tid; j < Hd; j += 256) p += attW[j] * tanhf(ppr[j] + crr[j]);
    red[tid] = p; __syncthreads();
    for (int s = 128; s > 0; s >>= 1) { if (tid < s) red[tid] += red[tid + s]; __syncthreads(); }
    if (tid == 0) w[t] = (t < lim) ? (red[0] + attB[0]) : -3.0e38f;
    __syncthreads();
  }
  if (tid == 0) {
    float m = -3.0e38f;
    for (int t = 0; t < L; ++t) m = fmaxf(m, w[t]);
    float s = 0.0f;
    for (int t = 0; t < L; ++t) { float e = __expf(w[t] - m); w[t] = e; s += e; }
    float inv = 1.0f / s;
    for (int t = 0; t < L; ++t) w[t] *= inv;
  }
  __syncthreads();
  float ac[4];
  float gp = 0.0f;
#pragma unroll
  for (int q = 0; q < 4; ++q) {
    int j = tid + q * 256;
    float a = 0.0f;
    for (int t = 0; t < L; ++t) a += w[t] * bf2f(hall1[((size_t)(t * 32 + b)) * Hd + j]);
    ac[q] = a;
    gp += gateW[j] * h2[(size_t)b * Hd + j] + gateW[Hd + j] * a;
  }
  red[tid] = gp; __syncthreads();
  for (int s = 128; s > 0; s >>= 1) { if (tid < s) red[tid] += red[tid + s]; __syncthreads(); }
  if (tid == 0) gsh = sigm(red[0] + gateB[0]);
  __syncthreads();
  float g = gsh;
#pragma unroll
  for (int q = 0; q < 4; ++q) {
    int j = tid + q * 256;
    size_t idx = (size_t)b * Hd + j;
    c2[idx] = g * c2[idx] + (1.0f - g) * ac[q];
  }
}

// ---------- launch helpers ----------
static inline dim3 g1(long long n) { return dim3((unsigned)((n + 255) / 256)); }
static inline void launch_gemm(hipStream_t s, const bf16* A, const bf16* Bp, const float* bias,
                               float* C, bf16* Cbf, int M, int N, int K, int act, int permM) {
  int tiles = (M >> 4) * (N >> 4);
  int blocks = (tiles + 7) >> 3;
  gemm_bf16_wmma<<<blocks, 256, 0, s>>>(A, Bp, bias, C, Cbf, M, N, K, act, permM);
}
static inline void launch_gemm_big(hipStream_t s, const bf16* A, const bf16* Bp, const float* bias,
                                   float* C, int Mpad, int Mreal, int N, int K, int permM) {
  dim3 grid((unsigned)(N >> 6), (unsigned)(Mpad >> 7));
  gemm_bf16_wmma_big<<<grid, 256, 0, s>>>(A, Bp, bias, C, Mreal, N, K, permM);
}

extern "C" void kernel_launch(void* const* d_in, const int* in_sizes, int n_in,
                              void* d_out, int out_size, void* d_ws, size_t ws_size,
                              hipStream_t stream) {
  (void)in_sizes; (void)n_in; (void)out_size; (void)ws_size;
  const int Vv = 32000, E = 512, Hd = 1024, B = 32, T = 64, L = 63;
  const int Mreal = 2016, Mpad = 2048;   // 63*32 padded to 128-row macro-tiles

  const float* img_feature = (const float*)d_in[0];
  const int*   sent1       = (const int*)d_in[1];
  const int*   len1        = (const int*)d_in[2];
  const int*   sent2       = (const int*)d_in[3];
  const int*   len2        = (const int*)d_in[4];
  const float* embedding   = (const float*)d_in[5];
  const float* l1_Wih  = (const float*)d_in[6];
  const float* l1_Whh  = (const float*)d_in[7];
  const float* l1_bih  = (const float*)d_in[8];
  const float* l1_bhh  = (const float*)d_in[9];
  const float* l1_imgW = (const float*)d_in[10];
  const float* l1_imgb = (const float*)d_in[11];
  const float* l1_outW = (const float*)d_in[12];
  const float* l1_outb = (const float*)d_in[13];
  const float* l2_Wih  = (const float*)d_in[14];
  const float* l2_Whh  = (const float*)d_in[15];
  const float* l2_bih  = (const float*)d_in[16];
  const float* l2_bhh  = (const float*)d_in[17];
  const float* l2_imgW = (const float*)d_in[18];
  const float* l2_imgb = (const float*)d_in[19];
  const float* l2_outW = (const float*)d_in[20];
  const float* l2_outb = (const float*)d_in[21];
  const float* l2_curW = (const float*)d_in[22];
  const float* l2_curb = (const float*)d_in[23];
  const float* l2_prvW = (const float*)d_in[24];
  const float* l2_prvb = (const float*)d_in[25];
  const float* l2_attW = (const float*)d_in[26];
  const float* l2_attb = (const float*)d_in[27];
  const float* gateW   = (const float*)d_in[28];
  const float* gateB   = (const float*)d_in[29];

  // ---- workspace carve-up (256B aligned) ----
  size_t off = 0;
  auto alloc = [&](size_t bytes) -> void* {
    void* p = (void*)((char*)d_ws + off);
    off = (off + bytes + 255) & ~(size_t)255;
    return p;
  };
  bf16* WihT1p = (bf16*)alloc((size_t)4096 * 512 * 2);
  bf16* WhhT1p = (bf16*)alloc((size_t)4096 * 1024 * 2);
  bf16* OutT1p = (bf16*)alloc((size_t)Vv * 1024 * 2);
  bf16* ImgT1p = (bf16*)alloc((size_t)512 * 1536 * 2);
  bf16* WihT2p = (bf16*)alloc((size_t)4096 * 512 * 2);
  bf16* WhhT2p = (bf16*)alloc((size_t)4096 * 1024 * 2);
  bf16* OutT2p = (bf16*)alloc((size_t)Vv * 1024 * 2);
  bf16* ImgT2p = (bf16*)alloc((size_t)512 * 1536 * 2);
  bf16* CurTp  = (bf16*)alloc((size_t)1024 * 1024 * 2);
  bf16* PrevTp = (bf16*)alloc((size_t)1024 * 1024 * 2);
  float* bsum1 = (float*)alloc(4096 * 4);
  float* bsum2 = (float*)alloc(4096 * 4);
  bf16* imgA   = (bf16*)alloc((size_t)B * 1536 * 2);
  float* imgscr= (float*)alloc((size_t)B * E * 4);
  bf16* img1bf = (bf16*)alloc((size_t)B * E * 2);
  bf16* img2bf = (bf16*)alloc((size_t)B * E * 2);
  bf16* Aemb1  = (bf16*)alloc((size_t)Mpad * E * 2);       // padded to 2048 rows
  bf16* Aemb2  = (bf16*)alloc((size_t)Mpad * E * 2);
  float* Xg1   = (float*)alloc((size_t)64 * B * 4096 * 4);
  float* Xg2   = (float*)alloc((size_t)64 * B * 4096 * 4);
  float* Hg    = (float*)alloc((size_t)B * 4096 * 4);
  float* c1    = (float*)alloc((size_t)B * Hd * 4);
  float* h1f   = (float*)alloc((size_t)B * Hd * 4);
  bf16*  h1bf  = (bf16*)alloc((size_t)B * Hd * 2);
  float* c2    = (float*)alloc((size_t)B * Hd * 4);
  float* h2f   = (float*)alloc((size_t)B * Hd * 4);
  bf16*  h2bf  = (bf16*)alloc((size_t)B * Hd * 2);
  bf16* Hall1  = (bf16*)alloc((size_t)Mpad * Hd * 2);      // padded
  bf16* Hall2  = (bf16*)alloc((size_t)Mpad * Hd * 2);
  float* pp    = (float*)alloc((size_t)L * B * Hd * 4);
  float* curb  = (float*)alloc((size_t)B * Hd * 4);

  float* out = (float*)d_out;
  const size_t PRB = (size_t)B * L * Vv;
  float* out_l1   = out;
  float* out_s1   = out + PRB;
  float* out_len1 = out + PRB + (size_t)B * T;
  float* out_l2   = out + PRB + (size_t)B * T + B;
  float* out_s2   = out_l2 + PRB;
  float* out_len2 = out_s2 + (size_t)B * T;

  // ---- pack weights to bf16 WMMA B-fragments ----
  pack_w<<<g1((long long)4096 * 512), 256, 0, stream>>>(l1_Wih, WihT1p, 4096, 512);
  pack_w<<<g1((long long)4096 * 1024), 256, 0, stream>>>(l1_Whh, WhhT1p, 4096, 1024);
  pack_w<<<g1((long long)Vv * 1024), 256, 0, stream>>>(l1_outW, OutT1p, Vv, 1024);
  pack_w<<<g1((long long)512 * 1536), 256, 0, stream>>>(l1_imgW, ImgT1p, 512, 1536);
  pack_w<<<g1((long long)4096 * 512), 256, 0, stream>>>(l2_Wih, WihT2p, 4096, 512);
  pack_w<<<g1((long long)4096 * 1024), 256, 0, stream>>>(l2_Whh, WhhT2p, 4096, 1024);
  pack_w<<<g1((long long)Vv * 1024), 256, 0, stream>>>(l2_outW, OutT2p, Vv, 1024);
  pack_w<<<g1((long long)512 * 1536), 256, 0, stream>>>(l2_imgW, ImgT2p, 512, 1536);
  pack_w<<<g1((long long)1024 * 1024), 256, 0, stream>>>(l2_curW, CurTp, 1024, 1024);
  pack_w<<<g1((long long)1024 * 1024), 256, 0, stream>>>(l2_prvW, PrevTp, 1024, 1024);
  vec_add2<<<g1(4096), 256, 0, stream>>>(l1_bih, l1_bhh, bsum1, 4096);
  vec_add2<<<g1(4096), 256, 0, stream>>>(l2_bih, l2_bhh, bsum2, 4096);
  f32_to_bf16<<<g1(B * 1536), 256, 0, stream>>>(img_feature, imgA, B * 1536);
  embed_gather<<<g1((long long)L * B * E), 256, 0, stream>>>(embedding, sent1, Aemb1, L * B * E);
  embed_gather<<<g1((long long)L * B * E), 256, 0, stream>>>(embedding, sent2, Aemb2, L * B * E);
  // zero the pad rows of the padded A operands
  zero_bf16<<<g1((Mpad - Mreal) * E), 256, 0, stream>>>(Aemb1 + (size_t)Mreal * E, (Mpad - Mreal) * E);
  zero_bf16<<<g1((Mpad - Mreal) * E), 256, 0, stream>>>(Aemb2 + (size_t)Mreal * E, (Mpad - Mreal) * E);
  zero_bf16<<<g1((Mpad - Mreal) * Hd), 256, 0, stream>>>(Hall1 + (size_t)Mreal * Hd, (Mpad - Mreal) * Hd);
  zero_bf16<<<g1((Mpad - Mreal) * Hd), 256, 0, stream>>>(Hall2 + (size_t)Mreal * Hd, (Mpad - Mreal) * Hd);

  // ================= layer 1 =================
  launch_gemm(stream, imgA, ImgT1p, l1_imgb, imgscr, img1bf, 32, 512, 1536, 1, 0);
  launch_gemm(stream, img1bf, WihT1p, bsum1, Xg1, nullptr, 32, 4096, 512, 0, 0);
  launch_gemm_big(stream, Aemb1, WihT1p, bsum1, Xg1 + (size_t)32 * 4096, Mpad, Mreal, 4096, 512, 0);
  zero_bf16<<<g1(B * Hd), 256, 0, stream>>>(h1bf, B * Hd);
  zero_f32<<<g1(B * Hd), 256, 0, stream>>>(c1, B * Hd);
  for (int s = 0; s < 64; ++s) {
    launch_gemm(stream, h1bf, WhhT1p, nullptr, Hg, nullptr, 32, 4096, 1024, 0, 0);
    lstm_elem<<<g1(B * Hd), 256, 0, stream>>>(Xg1 + (size_t)s * B * 4096, Hg, c1, h1f, h1bf,
                                              s ? Hall1 + (size_t)(s - 1) * B * Hd : (bf16*)nullptr);
  }
  launch_gemm_big(stream, Hall1, OutT1p, l1_outb, out_l1, Mpad, Mreal, Vv, 1024, L);
  launch_gemm_big(stream, Hall1, PrevTp, l2_prvb, pp, Mpad, Mreal, 1024, 1024, 0);

  // ================= layer 2 =================
  launch_gemm(stream, imgA, ImgT2p, l2_imgb, imgscr, img2bf, 32, 512, 1536, 1, 0);
  launch_gemm(stream, img2bf, WihT2p, bsum2, Xg2, nullptr, 32, 4096, 512, 0, 0);
  launch_gemm_big(stream, Aemb2, WihT2p, bsum2, Xg2 + (size_t)32 * 4096, Mpad, Mreal, 4096, 512, 0);
  zero_bf16<<<g1(B * Hd), 256, 0, stream>>>(h2bf, B * Hd);
  zero_f32<<<g1(B * Hd), 256, 0, stream>>>(c2, B * Hd);
  for (int s = 0; s < 64; ++s) {
    if (s) {
      launch_gemm(stream, h2bf, CurTp, l2_curb, curb, nullptr, 32, 1024, 1024, 0, 0);
      attn_step<<<32, 256, 0, stream>>>(Hall1, pp, curb, l2_attW, l2_attb, gateW, gateB, len1, h2f, c2);
    }
    launch_gemm(stream, h2bf, WhhT2p, nullptr, Hg, nullptr, 32, 4096, 1024, 0, 0);
    lstm_elem<<<g1(B * Hd), 256, 0, stream>>>(Xg2 + (size_t)s * B * 4096, Hg, c2, h2f, h2bf,
                                              s ? Hall2 + (size_t)(s - 1) * B * Hd : (bf16*)nullptr);
  }
  launch_gemm_big(stream, Hall2, OutT2p, l2_outb, out_l2, Mpad, Mreal, Vv, 1024, L);

  // ---- pass-through integer outputs ----
  i2f<<<g1(B * T), 256, 0, stream>>>(sent1, out_s1, B * T);
  i2f<<<g1(B), 256, 0, stream>>>(len1, out_len1, B);
  i2f<<<g1(B * T), 256, 0, stream>>>(sent2, out_s2, B * T);
  i2f<<<g1(B), 256, 0, stream>>>(len2, out_len2, B);
}
